// ScaledDotProductAttention_24378234372209
// MI455X (gfx1250) — compile-verified, matrix operand order checked by reading, not run on hardware
//
#include <hip/hip_runtime.h>
#include <hip/hip_bf16.h>

typedef __attribute__((ext_vector_type(16))) _Float16 v16h;
typedef __attribute__((ext_vector_type(8)))  float    v8f;

namespace {
constexpr int kH  = 8;
constexpr int kS  = 256;
constexpr int kDK = 64;
constexpr int kC  = 128;                 // window = 2*cutoff
constexpr float kScale = 0.125f;         // 1/sqrt(64)
constexpr unsigned kFragHalfs = 16u * 1024u * 1024u;  // 8*16*128*2*32*16 halfs per matrix
}

// Windowed-context gather, fully unwound from the reference's reshape/flip/pad
// chain.  src is the flat [H,S,DK] input; returns element Xc[h,row,wc,d].
__device__ __forceinline__ float con_val(const float* __restrict__ src,
                                         int h, int row, int wc, int d) {
    const int w  = 192 + wc;                 // slice [S-c : S+c) of the 2S axis
    const int i  = 32 * h + (row >> 3);
    const int jj = ((row & 7) << 6) + (w >> 3);
    const int dd = ((w & 7) << 6) + d;
    if (jj < kS) {                           // "up" half: i>=1 && jj >= S-i
        return (i >= 1 && jj >= kS - i) ? src[jj * 512 + dd] : 0.f;
    }
    const int j2 = jj - kS;                  // "low" half: j2 <= S-1-i
    return (j2 <= kS - 1 - i) ? src[j2 * 512 + dd] : 0.f;
}

// Stage 1: gather Q/K windows into f16 buffers laid out in exact WMMA fragment
// order: [h][tile16][c][half(2)][lane(32)][16 halfs].  A-matrix 16x32 layout:
// lane<16: M=lane, K = {0..7, 16..23};  lane>=16: M=lane-16, K = {8..15, 24..31}.
// B fragment uses the mirrored layout with N in place of M.
__global__ void __launch_bounds__(256) build_frags(const float* __restrict__ Q,
                                                   const float* __restrict__ K,
                                                   _Float16* __restrict__ qfrag,
                                                   _Float16* __restrict__ kfrag) {
    unsigned t = blockIdx.x * 256u + threadIdx.x;     // 2,097,152 threads
    const int lane = t & 31;  t >>= 5;
    const int half = t & 1;   t >>= 1;
    const int c    = t & 127; t >>= 7;
    const int qt   = t & 15;  t >>= 4;
    const int h    = t & 7;   t >>= 3;
    const int which = (int)t;                          // 0 = Q(A), 1 = K(B)

    const int row = qt * 16 + (lane & 15);
    const int w   = 192 + c;
    const int i   = 32 * h + (row >> 3);
    const int jj  = ((row & 7) << 6) + (w >> 3);
    bool ok; int sidx;
    if (jj < kS) { sidx = jj;       ok = (i >= 1) && (jj >= kS - i); }
    else         { sidx = jj - kS;  ok = (sidx <= kS - 1 - i); }

    const float* src  = which ? K : Q;
    // d = half*32 + K_local ; K_local = base_k + (lane<16 ? 0 : 8)
    const float* srow = src + sidx * 512 + ((w & 7) << 6) + half * 32
                            + ((lane < 16) ? 0 : 8);
    _Float16 tmp[16];
    if (ok) {
#pragma unroll
        for (int v = 0; v < 8; ++v) tmp[v]     = (_Float16)srow[v];       // K 0..7
#pragma unroll
        for (int v = 0; v < 8; ++v) tmp[8 + v] = (_Float16)srow[16 + v];  // K 16..23
    } else {
#pragma unroll
        for (int v = 0; v < 16; ++v) tmp[v] = (_Float16)0.f;
    }
    _Float16* dst = (which ? kfrag : qfrag)
        + ((unsigned)((((h * 16 + qt) * 128 + c) * 2 + half) * 32 + lane)) * 16u;
    *(uint4*)(dst)     = *(uint4*)(tmp);
    *(uint4*)(dst + 8) = *(uint4*)(tmp + 8);
}

// Ksum[h][c][d] = sum_s Kc[h,s,c,d]  (f32, exact)
__global__ void __launch_bounds__(256) build_ksum(const float* __restrict__ K,
                                                  float* __restrict__ ksum) {
    const int t = blockIdx.x * 256 + threadIdx.x;      // 65,536 = 8*128*64
    const int d = t & 63, c = (t >> 6) & 127, h = t >> 13;
    float s = 0.f;
    for (int row = 0; row < kS; ++row) s += con_val(K, h, row, c, d);
    ksum[t] = s;                                        // (h*128 + c)*64 + d
}

// Stage 2: one block per (h, q-tile, k-tile).  128 c-slices of the 16x16 score
// tile via v_wmma_f32_16x16x32_f16 (K=64 -> 2 chained WMMAs), parked in LDS,
// then an in-LDS softmax over c.  Writes attn_out (==1.0 row-sums) and the
// per-kt partial A[h,q,c] sums (deterministic: shfl_xor pair-combine, no atomics).
__global__ void __launch_bounds__(32) attn_scores_softmax(
        const _Float16* __restrict__ qfrag,
        const _Float16* __restrict__ kfrag,
        const unsigned char* __restrict__ mask,   // all-false in harness
        float* __restrict__ Apart,                // [H,S,16 kt,128 c]
        float* __restrict__ out_attn)             // [H,S,S]
{
    extern __shared__ float sc[];                 // 128 c * 256 f32 = 128 KB
    const int lane = threadIdx.x;
    int b = blockIdx.x;
    const int kt = b & 15;  b >>= 4;
    const int qt = b & 15;  b >>= 4;
    const int h  = b;

    const v16h* qv = (const v16h*)qfrag;
    const v16h* kv = (const v16h*)kfrag;
    const int fq = (h * 16 + qt) * 8192;          // v16h units: *128c*64
    const int fk = (h * 16 + kt) * 8192;

    // ---- Phase A: WMMA (EXEC all-ones: full wave, no divergence) ----
    for (int c = 0; c < kC; ++c) {
        v16h a0 = qv[fq + c * 64 + lane];
        v16h a1 = qv[fq + c * 64 + 32 + lane];
        v16h b0 = kv[fk + c * 64 + lane];
        v16h b1 = kv[fk + c * 64 + 32 + lane];
        v8f acc = {};
        acc = __builtin_amdgcn_wmma_f32_16x16x32_f16(false, a0, false, b0,
                                                     (short)0, acc, false, false);
        acc = __builtin_amdgcn_wmma_f32_16x16x32_f16(false, a1, false, b1,
                                                     (short)0, acc, false, false);
        // D layout: VGPR r, lane l -> (M = r + (l<16?0:8), N = l%16).
        // Store lane-major so each lane's 8 f32 are contiguous (2x ds_store_b128).
        *(v8f*)(sc + (c * 32 + lane) * 8) = acc;
    }
    __syncthreads();

    // ---- Phase B: softmax over c for this wave's 256 (q,k) pairs ----
    const int m  = lane & 15;          // q row within tile
    const int nh = lane >> 4;          // which 8-column half of k
    const int r  = m & 7;
    const int lbase = (m < 8) ? 0 : 16;
    const int q  = qt * 16 + m;
    float invZ[8];
    for (int nn = 0; nn < 8; ++nn) {
        const int n  = nh * 8 + nn;
        const int kg = kt * 16 + n;
        float* col = sc + (n + lbase) * 8 + r;          // stride 256 f32 per c
        const bool msk = mask[(h * 256 + q) * 256 + kg] != 0;
        float mx = -3.0e38f;
        for (int c = 0; c < kC; ++c) {
            float s = msk ? -1.0e9f : kScale * col[c * 256];
            col[c * 256] = s;
            mx = fmaxf(mx, s);
        }
        float Z = 0.f;
        for (int c = 0; c < kC; ++c) {
            float e = __expf(col[c * 256] - mx);
            col[c * 256] = e;                            // keep exp for A-accum
            Z += e;
        }
        invZ[nn] = 1.f / Z;
        out_attn[(h * 256 + q) * 256 + kg] = 1.0f;       // softmax row-sum == 1
    }
    // A-partials: t(lane) = sum over this lane's 8 k-columns of attn[q,k,c];
    // combine the two nh-halves commutatively via shfl_xor -> deterministic.
    float* Arow = Apart + ((h * 256 + q) * 16 + kt) * 128;
    for (int c = 0; c < kC; ++c) {
        const float* rowc = sc + c * 256 + lbase * 8 + r;
        float t = 0.f;
#pragma unroll
        for (int nn = 0; nn < 8; ++nn) t += rowc[(nh * 8 + nn) * 8] * invZ[nn];
        t += __shfl_xor(t, 16, 32);                      // a+b == b+a exactly
        if (nh == 0) Arow[c] = t;
    }
}

// Asum[h,q,c] = sum_kt Apart  (fixed order -> deterministic)
__global__ void __launch_bounds__(256) reduce_A(const float* __restrict__ Apart,
                                                float* __restrict__ Asum) {
    const int t = blockIdx.x * 256 + threadIdx.x;        // 262,144 = 8*256*128
    const int c = t & 127, hq = t >> 7;
    const float* p = Apart + hq * (16 * 128) + c;
    float s = 0.f;
#pragma unroll
    for (int kt = 0; kt < 16; ++kt) s += p[kt * 128];
    Asum[t] = s;
}

// context[h,q,d] = sum_c Asum[h,q,c] * Ksum[h,c,d]
__global__ void __launch_bounds__(256) context_kernel(const float* __restrict__ Asum,
                                                      const float* __restrict__ ksum,
                                                      float* __restrict__ out_ctx) {
    const int t = blockIdx.x * 256 + threadIdx.x;        // 131,072 = 8*256*64
    const int d = t & 63;
    const int q = (t >> 6) & 255;
    const int h = t >> 14;
    const float* Ar = Asum + (h * 256 + q) * 128;
    const float* Ks = ksum + h * 8192 + d;
    float acc = 0.f;
#pragma unroll 8
    for (int c = 0; c < kC; ++c) acc += Ar[c] * Ks[c * 64];
    out_ctx[t] = acc;
}

extern "C" void kernel_launch(void* const* d_in, const int* in_sizes, int n_in,
                              void* d_out, int out_size, void* d_ws, size_t ws_size,
                              hipStream_t stream) {
    (void)in_sizes; (void)n_in; (void)out_size;
    const float* Q = (const float*)d_in[0];
    const float* K = (const float*)d_in[1];
    // d_in[2] (V) is dead in the reference ('con' branch overwrites V with K).
    const unsigned char* mask = (const unsigned char*)d_in[3];  // all-false

    // Workspace layout (~82 MiB total):
    _Float16* qfrag = (_Float16*)d_ws;                         // 32 MiB
    _Float16* kfrag = qfrag + kFragHalfs;                      // 32 MiB
    float*    ksum  = (float*)(kfrag + kFragHalfs);            // 256 KiB  [8,128,64]
    float*    Apart = ksum + 8 * 128 * 64;                     // 16 MiB   [8,256,16,128]
    float*    Asum  = Apart + 8 * 256 * 16 * 128;              // 1 MiB    [8,256,128]
    (void)ws_size;

    float* out_ctx  = (float*)d_out;                           // [1,8,256,64]
    float* out_attn = out_ctx + kH * kS * kDK;                 // [1,8,256,256]

    build_frags<<<8192, 256, 0, stream>>>(Q, K, qfrag, kfrag);
    build_ksum<<<256, 256, 0, stream>>>(K, ksum);
    attn_scores_softmax<<<kH * 16 * 16, 32, kC * 256 * sizeof(float), stream>>>(
        qfrag, kfrag, mask, Apart, out_attn);
    reduce_A<<<1024, 256, 0, stream>>>(Apart, Asum);
    context_kernel<<<512, 256, 0, stream>>>(Asum, ksum, out_ctx);
}
